// GraphRNNEncoder_19954418057811
// MI455X (gfx1250) — matrix operand order repaired and checked by reading, not compiled, not run on hardware
//
#include <hip/hip_runtime.h>

#define NNODES 256
#define NF 64
#define NB 8

typedef __bf16 bf16_t;
typedef bf16_t v16bf __attribute__((ext_vector_type(16)));
typedef float  v8f   __attribute__((ext_vector_type(8)));

struct U4 { unsigned int a, b, c, d; };

union Frag {
  bf16_t h[16];
  U4 q[2];
  v16bf v;
};

__device__ __forceinline__ bf16_t f2bf(float f) { return (bf16_t)f; }
__device__ __forceinline__ float  bf2f(bf16_t h) { return (float)h; }

// ---------------------------------------------------------------------------
// Precompute W2 as WMMA B-matrix fragments (bf16), lane-ordered:
//   frag fi = nt*2+kc; entry (fi, lane) = 16 bf16 = 32B
//   B[g,f] = W2[f,g]; lane l covers N-col = 16*nt+(l&15), K = 32*kc+8*(l>>4)+(e&7)+16*(e>>3)
// Table = 8 frags * 32 lanes * 32B = 8KB, L1/L2 resident for the whole run.
// ---------------------------------------------------------------------------
__global__ void __launch_bounds__(256)
w2prep_kernel(const float* __restrict__ W2, bf16_t* __restrict__ w2frag)
{
  const int t  = threadIdx.x;   // 256 threads = 8 frags * 32 lanes
  const int fi = t >> 5;
  const int l  = t & 31;
  const int nt = fi >> 1;
  const int kc = fi & 1;
  const float* src = W2 + (size_t)(nt * 16 + (l & 15)) * NF + kc * 32 + ((l >> 4) << 3);
  bf16_t* dst = w2frag + ((size_t)fi * 32 + l) * 16;
#pragma unroll
  for (int e = 0; e < 16; ++e) dst[e] = f2bf(src[(e & 7) + ((e >> 3) << 4)]);
}

// ---------------------------------------------------------------------------
// Step 1 (messages==0): base = x @ W1^T ; msg[i,j,:] = relu(base[i,:]) * A[i,j]
// ---------------------------------------------------------------------------
__global__ void __launch_bounds__(256)
init_kernel(const float* __restrict__ x, const int* __restrict__ adj,
            const float* __restrict__ W1, float* __restrict__ base,
            bf16_t* __restrict__ msg_out)
{
  __shared__ float xrow[NF];
  __shared__ float brelu[NF];
  const int tid = threadIdx.x;
  const int b = blockIdx.x / NNODES;
  const int i = blockIdx.x % NNODES;

  if (tid < NF) xrow[tid] = x[((size_t)b * NNODES + i) * NF + tid];
  __syncthreads();
  if (tid < NF) {
    const float* w = W1 + tid * NF;
    float acc = 0.f;
#pragma unroll 8
    for (int g = 0; g < NF; ++g) acc += w[g] * xrow[g];
    base[((size_t)b * NNODES + i) * NF + tid] = acc;
    brelu[tid] = fmaxf(acc, 0.f);
  }
  __syncthreads();

  const float a = (float)adj[((size_t)b * NNODES + i) * NNODES + tid];
  bf16_t* orow = msg_out + (((size_t)b * NNODES + i) * NNODES + tid) * NF;
#pragma unroll
  for (int f0 = 0; f0 < NF; f0 += 8) {
    Frag fr;
#pragma unroll
    for (int e = 0; e < 8; ++e) fr.h[e] = f2bf(brelu[f0 + e] * a);
    *(U4*)(orow + f0) = fr.q[0];
  }
}

// ---------------------------------------------------------------------------
// One message-passing step for column i of batch b.
//   Mcol[k,g] = m[k,i,g]                        (LDS, bf16)
//   WM[k,f]   = sum_g Mcol[k,g] * W2[f,g]       (WMMA bf16 -> f32, stored bf16)
//   S[f]      = sum_k A[i,k] * WM[k,f]
//   new[i,j,f]= relu(base[i,f] + S[f] - WM[j,f]) * A[i,j]
// ---------------------------------------------------------------------------
__global__ void __launch_bounds__(256)
step_kernel(const bf16_t* __restrict__ msg_in,
            const int* __restrict__ adj,
            const float* __restrict__ base,
            const bf16_t* __restrict__ w2frag,
            bf16_t* __restrict__ msg_out)
{
  __shared__ __align__(16) bf16_t colbuf[NNODES * NF]; // 32KB: Mcol, then WM
  __shared__ float adjf[NNODES];
  __shared__ float red[256];
  __shared__ float Svec[NF];
  __shared__ float basef[NF];

  const int tid  = threadIdx.x;
  const int lane = tid & 31;
  const int wv   = tid >> 5;
  const int b = blockIdx.x / NNODES;
  const int i = blockIdx.x % NNODES;

  // Load column i of messages: thread tid -> row k = tid (128B contiguous)
  {
    const U4* src = (const U4*)(msg_in + (((size_t)b * NNODES + tid) * NNODES + i) * NF);
    U4* dst = (U4*)(colbuf + tid * NF);
#pragma unroll
    for (int q = 0; q < 8; ++q) dst[q] = src[q];
  }
  adjf[tid] = (float)adj[((size_t)b * NNODES + i) * NNODES + tid];
  if (tid < NF) basef[tid] = base[((size_t)b * NNODES + i) * NF + tid];
  __syncthreads();

  // --- GEMM: wave wv owns rows [32*wv, 32*wv+32) (reads & writes only these)
  {
    const int hi8 = (lane >> 4) << 3;   // A-matrix per-lane K sub-offset
    const int m0 = 32 * wv + (lane & 15);
    const int m1 = m0 + 16;

    Frag a00, a01, a10, a11;            // (row-tile, K-chunk)
    a00.q[0] = *(const U4*)(colbuf + m0 * NF +  0 + hi8);
    a00.q[1] = *(const U4*)(colbuf + m0 * NF + 16 + hi8);
    a01.q[0] = *(const U4*)(colbuf + m0 * NF + 32 + hi8);
    a01.q[1] = *(const U4*)(colbuf + m0 * NF + 48 + hi8);
    a10.q[0] = *(const U4*)(colbuf + m1 * NF +  0 + hi8);
    a10.q[1] = *(const U4*)(colbuf + m1 * NF + 16 + hi8);
    a11.q[0] = *(const U4*)(colbuf + m1 * NF + 32 + hi8);
    a11.q[1] = *(const U4*)(colbuf + m1 * NF + 48 + hi8);

    const int ncol = lane & 15;
    const int r0 = 32 * wv + ((lane >> 4) << 3); // C/D row base for this lane
    const U4* ftab = (const U4*)w2frag;          // 2 U4 per (frag,lane) entry
#pragma unroll
    for (int nt = 0; nt < 4; ++nt) {
      v8f acc0 = {};
      v8f acc1 = {};
#pragma unroll
      for (int kc = 0; kc < 2; ++kc) {
        Frag fb;                                 // precomputed B fragment
        const U4* fp = ftab + (((size_t)(nt * 2 + kc) * 32) + lane) * 2;
        fb.q[0] = fp[0];
        fb.q[1] = fp[1];
        acc0 = __builtin_amdgcn_wmma_f32_16x16x32_bf16(
            false, (kc ? a01.v : a00.v), false, fb.v, (short)0, acc0, false, false);
        acc1 = __builtin_amdgcn_wmma_f32_16x16x32_bf16(
            false, (kc ? a11.v : a10.v), false, fb.v, (short)0, acc1, false, false);
      }
#pragma unroll
      for (int r = 0; r < 8; ++r) {   // in-place WM write (wave-local rows)
        colbuf[(r0 + r)      * NF + nt * 16 + ncol] = f2bf(acc0[r]);
        colbuf[(r0 + 16 + r) * NF + nt * 16 + ncol] = f2bf(acc1[r]);
      }
    }
  }
  __syncthreads();

  // --- S[f] = sum_k adj[i,k] * WM[k,f]  (4 partial chunks of 64 k each)
  {
    const int f = tid & 63;
    const int c = tid >> 6;
    float p = 0.f;
    const int k0 = c * 64;
#pragma unroll 8
    for (int k = 0; k < 64; ++k) p += adjf[k0 + k] * bf2f(colbuf[(k0 + k) * NF + f]);
    red[tid] = p;
    __syncthreads();
    if (tid < 64) Svec[tid] = red[tid] + red[tid + 64] + red[tid + 128] + red[tid + 192];
    __syncthreads();
  }

  // --- emit new message row j = tid
  {
    const float a = adjf[tid];
    bf16_t* orow = msg_out + (((size_t)b * NNODES + i) * NNODES + tid) * NF;
#pragma unroll
    for (int f0 = 0; f0 < NF; f0 += 8) {
      Frag fr;
#pragma unroll
      for (int e = 0; e < 8; ++e) {
        float v = basef[f0 + e] + Svec[f0 + e] - bf2f(colbuf[tid * NF + f0 + e]);
        fr.h[e] = f2bf(fmaxf(v, 0.f) * a);
      }
      *(U4*)(orow + f0) = fr.q[0];
    }
  }
}

// ---------------------------------------------------------------------------
// enc[i,f] = relu( (Un@x)[i,f] + Um @ (sum_j A[i,j] m[j,i,:]) )
// ---------------------------------------------------------------------------
__global__ void __launch_bounds__(256)
agg_kernel(const bf16_t* __restrict__ msg,
           const int* __restrict__ adj,
           const float* __restrict__ x,
           const float* __restrict__ Un,
           const float* __restrict__ Um,
           float* __restrict__ enc)
{
  __shared__ __align__(16) bf16_t colbuf[NNODES * NF];
  __shared__ float adjf[NNODES];
  __shared__ float red[256];
  __shared__ float Tvec[NF];
  const int tid = threadIdx.x;
  const int b = blockIdx.x / NNODES;
  const int i = blockIdx.x % NNODES;

  {
    const U4* src = (const U4*)(msg + (((size_t)b * NNODES + tid) * NNODES + i) * NF);
    U4* dst = (U4*)(colbuf + tid * NF);
#pragma unroll
    for (int q = 0; q < 8; ++q) dst[q] = src[q];
  }
  adjf[tid] = (float)adj[((size_t)b * NNODES + i) * NNODES + tid];
  __syncthreads();

  const int f = tid & 63;
  const int c = tid >> 6;

  float p = 0.f;
#pragma unroll 8
  for (int j = 0; j < 64; ++j) p += adjf[c * 64 + j] * bf2f(colbuf[(c * 64 + j) * NF + f]);
  red[tid] = p;
  __syncthreads();
  if (tid < 64) Tvec[tid] = red[tid] + red[tid + 64] + red[tid + 128] + red[tid + 192];
  __syncthreads();

  // Unx[f] = sum_j Un[i,j] * x[b,j,f]
  p = 0.f;
  const float* unrow = Un + (size_t)i * NNODES + c * 64;
  const float* xcol  = x + ((size_t)b * NNODES + c * 64) * NF + f;
#pragma unroll 8
  for (int j = 0; j < 64; ++j) p += unrow[j] * xcol[(size_t)j * NF];
  red[tid] = p;
  __syncthreads();
  if (tid < 64) {
    float unx = red[tid] + red[tid + 64] + red[tid + 128] + red[tid + 192];
    float ag = 0.f;
    const float* umrow = Um + tid * NF;
#pragma unroll 8
    for (int g = 0; g < NF; ++g) ag += umrow[g] * Tvec[g];
    enc[((size_t)b * NNODES + i) * NF + tid] = fmaxf(unx + ag, 0.f);
  }
}

// ---------------------------------------------------------------------------
// out[b,o] = sigmoid( enc[b,:] . lw[o,:] + lb[o] ), 16384-long dots, FC_OUT=8
// ---------------------------------------------------------------------------
__global__ void __launch_bounds__(256)
fc_kernel(const float* __restrict__ enc,
          const float* __restrict__ lw,
          const float* __restrict__ lb,
          float* __restrict__ out)
{
  __shared__ float red[256];
  const int tid = threadIdx.x;
  const int b = blockIdx.x;
  float p[8];
#pragma unroll
  for (int o = 0; o < 8; ++o) p[o] = 0.f;
  const float* e = enc + (size_t)b * (NNODES * NF);
  for (int idx = tid; idx < NNODES * NF; idx += 256) {
    const float ev = e[idx];
#pragma unroll
    for (int o = 0; o < 8; ++o) p[o] += ev * lw[(size_t)o * (NNODES * NF) + idx];
  }
  for (int o = 0; o < 8; ++o) {
    red[tid] = p[o];
    __syncthreads();
    for (int s = 128; s > 0; s >>= 1) {
      if (tid < s) red[tid] += red[tid + s];
      __syncthreads();
    }
    if (tid == 0) {
      const float z = red[0] + lb[o];
      out[b * 8 + o] = 1.f / (1.f + __expf(-z));
    }
    __syncthreads();
  }
}

// ---------------------------------------------------------------------------
extern "C" void kernel_launch(void* const* d_in, const int* in_sizes, int n_in,
                              void* d_out, int out_size, void* d_ws, size_t ws_size,
                              hipStream_t stream)
{
  (void)in_sizes; (void)n_in; (void)out_size; (void)ws_size;
  const float* x   = (const float*)d_in[0];
  const int*   adj = (const int*)d_in[1];
  const float* W1  = (const float*)d_in[2];
  const float* W2  = (const float*)d_in[3];
  const float* Un  = (const float*)d_in[4];
  const float* Um  = (const float*)d_in[5];
  const float* lw  = (const float*)d_in[6];
  const float* lb  = (const float*)d_in[7];
  float* out = (float*)d_out;

  char* ws = (char*)d_ws;
  const size_t msg_elems = (size_t)NB * NNODES * NNODES * NF;   // 33.5M
  bf16_t* msgA = (bf16_t*)ws;                                    // 64 MB
  bf16_t* msgB = (bf16_t*)(ws + msg_elems * 2);                  // 64 MB
  float* base  = (float*)(ws + msg_elems * 4);                   // 512 KB
  float* enc   = base + (size_t)NB * NNODES * NF;                // 512 KB
  bf16_t* w2fr = (bf16_t*)(enc + (size_t)NB * NNODES * NF);      // 8 KB

  dim3 grid(NB * NNODES), block(256);
  // W2 -> lane-ordered bf16 B fragments (once)
  w2prep_kernel<<<dim3(1), block, 0, stream>>>(W2, w2fr);
  // t=1 (messages start at zero)
  init_kernel<<<grid, block, 0, stream>>>(x, adj, W1, base, msgA);
  // t=2..4, ping-pong buffers (kernel boundary = global sync between steps)
  step_kernel<<<grid, block, 0, stream>>>(msgA, adj, base, w2fr, msgB);
  step_kernel<<<grid, block, 0, stream>>>(msgB, adj, base, w2fr, msgA);
  step_kernel<<<grid, block, 0, stream>>>(msgA, adj, base, w2fr, msgB);
  // readout
  agg_kernel<<<grid, block, 0, stream>>>(msgB, adj, x, Un, Um, enc);
  fc_kernel<<<dim3(NB), block, 0, stream>>>(enc, lw, lb, out);
}
